// SRNN_scratch_14250701488640
// MI455X (gfx1250) — matrix operand order repaired and checked by reading, not compiled, not run on hardware
//
#include <hip/hip_runtime.h>
#include <hip/hip_bf16.h>

// ---------------------------------------------------------------------------
// SRNN on MI455X (gfx1250, wave32, WMMA).
//   Phase 0: convert weights f32->bf16, build shifted/transposed X, init h0.
//   Phase 1: WX0 = W[0] @ X          (dense bf16 WMMA GEMM, f32 accum)
//   Phase 2: persistent recurrent kernel, 512 steps x 3 layers.
//            Split grid barrier: independent U-matvecs overlap barrier waits;
//            only the W-chain sits on the critical path.
//   Phase 3: Y = V @ H2              (dense bf16 WMMA GEMM, f32 accum)
// ---------------------------------------------------------------------------

#define HD 2048
#define TT 512
#define LL 3
#define OO 2048

typedef __attribute__((ext_vector_type(16))) __bf16 v16bf;
typedef __attribute__((ext_vector_type(8)))  __bf16 v8bf;
typedef __attribute__((ext_vector_type(8)))  float  v8f;

// --- WMMA operand loaders (contiguous k order: k = 16*(lane/16) + s, A and B alike) -------

// A: 16x32 bf16 tile, row-major source with row stride ldk (elements).
__device__ inline v16bf load_a_tile(const __bf16* base, int ldk) {
  int lane = threadIdx.x & 31;
  int m    = lane & 15;
  int koff = (lane >> 4) << 4;
  const __bf16* p = base + (size_t)m * ldk + koff;
  v8bf lo = *(const v8bf*)(p);
  v8bf hi = *(const v8bf*)(p + 8);
  return __builtin_shufflevector(lo, hi, 0,1,2,3,4,5,6,7,8,9,10,11,12,13,14,15);
}

// B: 32x16 bf16 tile from a K-major (N rows x K cols) source, row stride ldk.
__device__ inline v16bf load_b_tile(const __bf16* base, int ldk) {
  int lane = threadIdx.x & 31;
  int n    = lane & 15;
  int koff = (lane >> 4) << 4;
  const __bf16* p = base + (size_t)n * ldk + koff;
  v8bf lo = *(const v8bf*)(p);
  v8bf hi = *(const v8bf*)(p + 8);
  return __builtin_shufflevector(lo, hi, 0,1,2,3,4,5,6,7,8,9,10,11,12,13,14,15);
}

// B for matvec: vector h broadcast into all 16 columns -> D columns replicated.
__device__ inline v16bf load_b_vec(const __bf16* hk0) {
  int lane = threadIdx.x & 31;
  int koff = (lane >> 4) << 4;
  const __bf16* p = hk0 + koff;
  v8bf lo = *(const v8bf*)(p);
  v8bf hi = *(const v8bf*)(p + 8);
  return __builtin_shufflevector(lo, hi, 0,1,2,3,4,5,6,7,8,9,10,11,12,13,14,15);
}

// --- Phase 0: convert / stage / init ---------------------------------------

__global__ void __launch_bounds__(256) convert_init(
    const float* __restrict__ inp, const float* __restrict__ U,
    const float* __restrict__ W,   const float* __restrict__ V,
    __bf16* __restrict__ Ubf, __bf16* __restrict__ Wbf, __bf16* __restrict__ Vbf,
    __bf16* __restrict__ Xt,  __bf16* __restrict__ hbuf, unsigned* __restrict__ bar)
{
  const size_t n1 = (size_t)LL * HD * HD;          // U, W element counts
  const size_t nV = (size_t)OO * HD;
  const size_t nX = (size_t)TT * HD;
  const size_t total = 2 * n1 + nV + nX + HD;
  for (size_t i = (size_t)blockIdx.x * blockDim.x + threadIdx.x; i < total;
       i += (size_t)gridDim.x * blockDim.x) {
    if (i < n1) {
      Ubf[i] = (__bf16)U[i];
    } else if (i < 2 * n1) {
      size_t j = i - n1;
      Wbf[j] = (__bf16)W[j];
    } else if (i < 2 * n1 + nV) {
      size_t j = i - 2 * n1;
      Vbf[j] = (__bf16)V[j];
    } else if (i < 2 * n1 + nV + nX) {
      size_t j = i - 2 * n1 - nV;                  // Xt[t][k] = input[k][t+1]
      int t = (int)(j / HD), k = (int)(j % HD);
      Xt[j] = (t + 1 < TT) ? (__bf16)inp[(size_t)k * TT + t + 1] : (__bf16)0.0f;
    } else {
      int k = (int)(i - 2 * n1 - nV - nX);         // h0 = input[:,0], parity 1
      __bf16 h = (__bf16)inp[(size_t)k * TT];
      hbuf[(0 * 2 + 1) * HD + k] = h;
      hbuf[(1 * 2 + 1) * HD + k] = h;
      hbuf[(2 * 2 + 1) * HD + k] = h;
    }
  }
  if (blockIdx.x == 0 && threadIdx.x == 0) { bar[0] = 0u; bar[1] = 0u; }
}

// --- Dense GEMM: C(MxN,f32) = A(MxK) * Bt(NxK)^T, all bf16 inputs ----------
// Wave computes a 16x64 tile (4 WMMA accumulators, A reuse x4).

__global__ void __launch_bounds__(256) gemm_bf16(
    const __bf16* __restrict__ A, const __bf16* __restrict__ Bt,
    float* __restrict__ C, int M, int N, int K)
{
  int wv = threadIdx.x >> 5;
  int nb = N >> 6;
  int bn = blockIdx.x % nb;
  int bm = blockIdx.x / nb;
  int m0 = (bm * 8 + wv) << 4;
  int n0 = bn << 6;
  v8f acc0 = {}, acc1 = {}, acc2 = {}, acc3 = {};
  const __bf16* pa = A + (size_t)m0 * K;
  const __bf16* pb = Bt + (size_t)n0 * K;
  for (int k0 = 0; k0 < K; k0 += 32) {
    __builtin_prefetch(pa + k0 + 256, 0, 3);       // global_prefetch (near)
    __builtin_prefetch(pb + k0 + 256, 0, 3);
    v16bf a  = load_a_tile(pa + k0, K);
    v16bf b0 = load_b_tile(pb + k0, K);
    v16bf b1 = load_b_tile(pb + (size_t)16 * K + k0, K);
    v16bf b2 = load_b_tile(pb + (size_t)32 * K + k0, K);
    v16bf b3 = load_b_tile(pb + (size_t)48 * K + k0, K);
    acc0 = __builtin_amdgcn_wmma_f32_16x16x32_bf16(false, a, false, b0, (short)0, acc0, false, false);
    acc1 = __builtin_amdgcn_wmma_f32_16x16x32_bf16(false, a, false, b1, (short)0, acc1, false, false);
    acc2 = __builtin_amdgcn_wmma_f32_16x16x32_bf16(false, a, false, b2, (short)0, acc2, false, false);
    acc3 = __builtin_amdgcn_wmma_f32_16x16x32_bf16(false, a, false, b3, (short)0, acc3, false, false);
  }
  // D layout: VGPR r, lanes 0-15 -> (M=r, N=lane); lanes 16-31 -> (M=8+r).
  int lane = threadIdx.x & 31;
  int col  = lane & 15;
  int rb   = (lane >> 4) << 3;
  v8f accs[4] = {acc0, acc1, acc2, acc3};
#pragma unroll
  for (int i = 0; i < 4; ++i)
#pragma unroll
    for (int r = 0; r < 8; ++r)
      C[(size_t)(m0 + rb + r) * N + (n0 + 16 * i + col)] = accs[i][r];
}

// --- Split grid barrier (monotonic generation counter) ----------------------

__device__ inline void bar_signal(unsigned* cnt, unsigned* gen, unsigned nblk) {
  __threadfence();
  __syncthreads();
  if (threadIdx.x == 0) {
    unsigned old = __hip_atomic_fetch_add(cnt, 1u, __ATOMIC_ACQ_REL, __HIP_MEMORY_SCOPE_AGENT);
    if (old == nblk - 1u) {
      __hip_atomic_store(cnt, 0u, __ATOMIC_RELAXED, __HIP_MEMORY_SCOPE_AGENT);
      __hip_atomic_fetch_add(gen, 1u, __ATOMIC_RELEASE, __HIP_MEMORY_SCOPE_AGENT);
    }
  }
}

__device__ inline void bar_wait(unsigned* gen, unsigned target) {
  if (threadIdx.x == 0) {
    while ((int)(__hip_atomic_load(gen, __ATOMIC_ACQUIRE, __HIP_MEMORY_SCOPE_AGENT) - target) < 0)
      __builtin_amdgcn_s_sleep(2);
  }
  __syncthreads();
  __threadfence();
}

// Accumulate C += Mrow0[16 x (kbeg..kend)] * h  via bf16 WMMA (h in all cols).
__device__ inline v8f matvec(const __bf16* Mrow0, const __bf16* h,
                             int kbeg, int kend, v8f c) {
  for (int k0 = kbeg; k0 < kend; k0 += 32) {
    v16bf a = load_a_tile(Mrow0 + k0, HD);
    v16bf b = load_b_vec(h + k0);
    c = __builtin_amdgcn_wmma_f32_16x16x32_bf16(false, a, false, b, (short)0, c, false, false);
  }
  return c;
}

// K-split x4 reduction in LDS, then bias+sigmoid, write bf16 h (and H2 row).
// Caller must follow with bar_signal (provides the __syncthreads separating
// the smem read here from the next phase's smem writes).
__device__ inline void reduce_act_store(const v8f& c, float* smem, int wv, int lane,
                                        int row0, const float* wx, int t,
                                        const float* bj, __bf16* hw, __bf16* h2row)
{
  if (lane == 0) {
#pragma unroll
    for (int r = 0; r < 8; ++r) smem[wv * 16 + r] = c[r];
  } else if (lane == 16) {
#pragma unroll
    for (int r = 0; r < 8; ++r) smem[wv * 16 + 8 + r] = c[r];
  }
  __syncthreads();
  if (((wv & 3) == 0) && lane < 16) {
    int row = row0 + lane;
    float s = smem[wv * 16 + lane] + smem[(wv + 1) * 16 + lane]
            + smem[(wv + 2) * 16 + lane] + smem[(wv + 3) * 16 + lane] + bj[row];
    if (wx) s += wx[(size_t)row * TT + t];
    float h = 1.0f / (1.0f + __expf(-s));
    hw[row] = (__bf16)h;
    if (h2row) h2row[row] = (__bf16)h;
  }
}

// --- Recurrent persistent kernel -------------------------------------------
// 64 blocks x 8 waves = 512 waves: 128 row-tiles (16 rows) x 4 K-quarters.

__global__ void __launch_bounds__(256) srnn_recurrent(
    const __bf16* __restrict__ Ubf, const __bf16* __restrict__ Wbf,
    const float* __restrict__ bias, const float* __restrict__ WX0,
    __bf16* __restrict__ hbuf, __bf16* __restrict__ H2, unsigned* __restrict__ bar)
{
  const unsigned NBLK = 64;
  int wv   = threadIdx.x >> 5;
  int lane = threadIdx.x & 31;
  int gw   = blockIdx.x * 8 + wv;      // 0..511 waves
  int tile = gw >> 2;                  // 0..127 : 16-row output tile
  int kq   = gw & 3;                   // K-split quarter
  int row0 = tile << 4;
  int kbeg = kq << 9;                  // 0,512,1024,1536
  int kend = kbeg + 512;
  __shared__ float smem[128];

  const size_t MS = (size_t)HD * HD;
  const __bf16* U0 = Ubf + (size_t)row0 * HD;
  const __bf16* U1 = Ubf + MS     + (size_t)row0 * HD;
  const __bf16* U2 = Ubf + 2 * MS + (size_t)row0 * HD;
  const __bf16* W1 = Wbf + MS     + (size_t)row0 * HD;
  const __bf16* W2 = Wbf + 2 * MS + (size_t)row0 * HD;
  unsigned* cnt = bar;
  unsigned* gen = bar + 1;
  unsigned nb = 0;

  // Prologue: U0 @ h0_initial (parity 1, written by convert_init).
  v8f c0 = {};
  c0 = matvec(U0, hbuf + (0 * 2 + 1) * HD, kbeg, kend, c0);

  for (int t = 0; t < TT; ++t) {
    int pr = (t & 1) ^ 1;              // read parity (previous step's h)
    int pw = t & 1;                    // write parity (this step's h)
    const __bf16* h1r = hbuf + (1 * 2 + pr) * HD;
    const __bf16* h2r = hbuf + (2 * 2 + pr) * HD;
    __bf16* h0w = hbuf + (0 * 2 + pw) * HD;
    __bf16* h1w = hbuf + (1 * 2 + pw) * HD;
    __bf16* h2w = hbuf + (2 * 2 + pw) * HD;

    // ---- layer 0: finish h0 = sigmoid(U0@h0_prev + WX0[.,t] + b0) ----
    reduce_act_store(c0, smem, wv, lane, row0, WX0, t, bias, h0w, nullptr);
    bar_signal(cnt, gen, NBLK); ++nb;

    // overlap barrier latency: independent U1 @ h1_prev
    v8f c1 = {};
    c1 = matvec(U1, h1r, kbeg, kend, c1);
    bar_wait(gen, nb);

    // ---- layer 1: h1 = sigmoid(U1@h1_prev + W1@h0 + b1) ----
    c1 = matvec(W1, h0w, kbeg, kend, c1);
    reduce_act_store(c1, smem, wv, lane, row0, nullptr, t, bias + HD, h1w, nullptr);
    bar_signal(cnt, gen, NBLK); ++nb;

    // overlap: independent U2 @ h2_prev
    v8f c2 = {};
    c2 = matvec(U2, h2r, kbeg, kend, c2);
    bar_wait(gen, nb);

    // ---- layer 2: h2 = sigmoid(U2@h2_prev + W2@h1 + b2); record H2[t] ----
    c2 = matvec(W2, h1w, kbeg, kend, c2);
    reduce_act_store(c2, smem, wv, lane, row0, nullptr, t, bias + 2 * HD, h2w,
                     H2 + (size_t)t * HD);
    bar_signal(cnt, gen, NBLK); ++nb;

    // overlap: next step's U0 @ h0 (h0w synced since this step's 1st barrier)
    c0 = v8f{};
    c0 = matvec(U0, h0w, kbeg, kend, c0);
    bar_wait(gen, nb);
  }
}

// --- Launcher ---------------------------------------------------------------

extern "C" void kernel_launch(void* const* d_in, const int* in_sizes, int n_in,
                              void* d_out, int out_size, void* d_ws, size_t ws_size,
                              hipStream_t stream) {
  (void)in_sizes; (void)n_in; (void)out_size; (void)ws_size;
  const float* inp = (const float*)d_in[0];   // (Hd, T)
  const float* U   = (const float*)d_in[1];   // (L, Hd, Hd)
  const float* W   = (const float*)d_in[2];   // (L, Hd, Hd)
  const float* b   = (const float*)d_in[3];   // (L, Hd)
  const float* V   = (const float*)d_in[4];   // (O, Hd)

  char* ws = (char*)d_ws;
  size_t o = 0;
  __bf16* Ubf  = (__bf16*)(ws + o); o += (size_t)LL * HD * HD * 2;  // 24 MB
  __bf16* Wbf  = (__bf16*)(ws + o); o += (size_t)LL * HD * HD * 2;  // 24 MB
  __bf16* Vbf  = (__bf16*)(ws + o); o += (size_t)OO * HD * 2;       //  8 MB
  __bf16* Xt   = (__bf16*)(ws + o); o += (size_t)TT * HD * 2;       //  2 MB
  __bf16* H2   = (__bf16*)(ws + o); o += (size_t)TT * HD * 2;       //  2 MB
  float*  WX0  = (float*) (ws + o); o += (size_t)HD * TT * 4;       //  4 MB
  __bf16* hbuf = (__bf16*)(ws + o); o += (size_t)LL * 2 * HD * 2;   // 24 KB
  o = (o + 255) & ~(size_t)255;
  unsigned* bar = (unsigned*)(ws + o);                              // {count, gen}

  // Phase 0: stage weights (bf16), shifted+transposed X, h0, barrier init.
  convert_init<<<2048, 256, 0, stream>>>(inp, U, W, V, Ubf, Wbf, Vbf, Xt, hbuf, bar);

  // Phase 1: WX0 (Hd x T, f32) = W[0](Hd x Hd) @ X(Hd x T)  [Bt = Xt, K-major]
  gemm_bf16<<<128, 256, 0, stream>>>(Wbf, Xt, WX0, HD, TT, HD);

  // Phase 2: 512-step persistent recurrence (64 blocks, split grid barrier).
  srnn_recurrent<<<64, 256, 0, stream>>>(Ubf, Wbf, b, WX0, hbuf, H2, bar);

  // Phase 3: Y (O x T, f32) = V(O x Hd) @ H2(Hd x T)  [Bt = H2, time-major]
  gemm_bf16<<<128, 256, 0, stream>>>(Vbf, H2, (float*)d_out, OO, TT, HD);
}
